// CumMaxLayer_15753940042280
// MI455X (gfx1250) — matrix operand order repaired and checked by reading, not compiled, not run on hardware
//
#include <hip/hip_runtime.h>
#include <stdint.h>

// ---- problem constants (from reference) ----
#define B_  16
#define H_  128
#define W_  128
#define C_  256            // output channels N
#define K_  512            // 2*C concat depth
#define M_  (B_ * H_ * W_) // 262144 rows

#define NTILE_ 64          // columns per block (one LDS-staged Wt slice)

typedef __bf16 v8bf  __attribute__((ext_vector_type(8)));
typedef __bf16 v16bf __attribute__((ext_vector_type(16)));
typedef float  v8f   __attribute__((ext_vector_type(8)));

static __device__ __forceinline__ uint16_t f32_to_bf16_bits(float f) {
    // round-to-nearest-even truncation to bf16
    uint32_t u = __float_as_uint(f);
    u += 0x7fffu + ((u >> 16) & 1u);
    return (uint16_t)(u >> 16);
}

// A-fragment loader: lane's row, 16 contiguous K (two adjacent 16B chunks)
static __device__ __forceinline__ v16bf load_a_frag(const uint16_t* p) {
    v8bf lo = *(const v8bf*)(p);
    v8bf hi = *(const v8bf*)(p + 8);
    return __builtin_shufflevector(lo, hi,
             0,1,2,3,4,5,6,7,8,9,10,11,12,13,14,15);
}

// ---------------------------------------------------------------------------
// Kernel 1: repack W [K_, C_] fp32 (row-major, k-major) -> Wt [C_, K_] bf16.
// Transposed layout makes each WMMA B-fragment one contiguous 32B chunk.
// ---------------------------------------------------------------------------
__global__ void repack_w_kernel(const float* __restrict__ Wsrc,
                                uint16_t* __restrict__ Wt) {
    const int k = blockIdx.x;   // 0..511
    const int n = threadIdx.x;  // 0..255
    Wt[(size_t)n * K_ + k] = f32_to_bf16_bits(Wsrc[(size_t)k * C_ + n]);
}

// ---------------------------------------------------------------------------
// Kernel 2: fused cummax + bf16 conversion into A [M_, K_] bf16 (row-major).
//   blockIdx.y == 0 : cummax along W  -> A[m, 0:256]
//   blockIdx.y == 1 : cummax along H  -> A[m, 256:512]
// grid is streamed once -> non-temporal loads (keep the 192MB L2 free for A,
// which is re-read by the GEMM and fits). A stores use default RT policy.
// ---------------------------------------------------------------------------
__global__ void cummax_kernel(const float* __restrict__ grid,
                              uint16_t* __restrict__ A) {
    const int c    = threadIdx.x;        // channel 0..255
    const int mode = blockIdx.y;         // 0: scan W, 1: scan H
    const int bb   = blockIdx.x >> 7;    // batch
    const int rc   = blockIdx.x & 127;   // h (mode 0) or w (mode 1)

    float run = -__FLT_MAX__;
    if (mode == 0) {
        const size_t mbase = ((size_t)bb * H_ + rc) * W_;
        const size_t gbase = mbase * C_;
        for (int w = 0; w < W_; ++w) {
            float v = __builtin_nontemporal_load(&grid[gbase + (size_t)w * C_ + c]);
            run = fmaxf(run, v);
            A[(mbase + w) * (size_t)K_ + c] = f32_to_bf16_bits(run);
        }
    } else {
        for (int h = 0; h < H_; ++h) {
            const size_t m = ((size_t)bb * H_ + h) * W_ + rc;
            float v = __builtin_nontemporal_load(&grid[m * (size_t)C_ + c]);
            run = fmaxf(run, v);
            A[m * (size_t)K_ + C_ + c] = f32_to_bf16_bits(run);
        }
    }
}

// ---------------------------------------------------------------------------
// Kernel 3: WMMA GEMM  out[M_, C_] = A[M_, K_] (bf16) x Wt[C_, K_]^T + bias.
// Block = 8 waves, each wave owns one 16-row M tile; ALL waves share one
// 64-column N group whose Wt slice (64x512 bf16 = 64KB) is staged in LDS.
//   - B fragments come from LDS (ds_load_b128, DScnt) -> broadcast reads,
//     8x less L2 traffic, and B waits no longer serialize the A load stream.
//   - A fragments stream from global (LOADcnt), likely L2-resident.
// Grid: (M/128, C/64) = (2048, 4).
// ---------------------------------------------------------------------------
__global__ __launch_bounds__(256)
void wmma_gemm_kernel(const uint16_t* __restrict__ A,
                      const uint16_t* __restrict__ Wt,
                      const float* __restrict__ bias,
                      float* __restrict__ out) {
    __shared__ uint16_t lds_w[NTILE_ * K_];   // 64KB: Wt slice [n0..n0+64) x K

    const int tid   = threadIdx.x;
    const int lane  = tid & 31;
    const int wave  = tid >> 5;               // 0..7 -> M tile within block

    const int tile_m = blockIdx.x * 128 + wave * 16;
    const int n0     = blockIdx.y * NTILE_;   // column group base

    // ---- cooperative LDS fill: Wt slice is a contiguous 64KB chunk ----
    {
        const uint4* src = (const uint4*)(Wt + (size_t)n0 * K_);
        uint4* dst = (uint4*)lds_w;
#pragma unroll
        for (int i = 0; i < (NTILE_ * K_ * 2) / 16 / 256; ++i)   // 16 chunks/thread
            dst[i * 256 + tid] = src[i * 256 + tid];
    }
    __syncthreads();

    const int lmod  = lane & 15;   // row (A) / col (B,C)
    const int lhalf = lane >> 4;   // which K/M half this lane covers

    // A fragment: lane row = tile_m+lmod; 16 contiguous K at k0 + lhalf*16.
    const uint16_t* arow = A + (size_t)(tile_m + lmod) * K_ + lhalf * 16;
    // B fragments from LDS: lane col = t*16+lmod; 16 contiguous K at k0+lhalf*16.
    const uint16_t* lb0 = lds_w + ( 0 + lmod) * K_ + lhalf * 16;
    const uint16_t* lb1 = lds_w + (16 + lmod) * K_ + lhalf * 16;
    const uint16_t* lb2 = lds_w + (32 + lmod) * K_ + lhalf * 16;
    const uint16_t* lb3 = lds_w + (48 + lmod) * K_ + lhalf * 16;

    v8f acc0 = {}, acc1 = {}, acc2 = {}, acc3 = {};

#pragma unroll
    for (int kt = 0; kt < K_ / 32; ++kt) {
        const int k0 = kt * 32;

        v16bf aF  = load_a_frag(arow + k0);
        v16bf bf0 = *(const v16bf*)(lb0 + k0);
        v16bf bf1 = *(const v16bf*)(lb1 + k0);
        v16bf bf2 = *(const v16bf*)(lb2 + k0);
        v16bf bf3 = *(const v16bf*)(lb3 + k0);

        acc0 = __builtin_amdgcn_wmma_f32_16x16x32_bf16(false, aF, false, bf0,
                                                       (short)0, acc0, false, false);
        acc1 = __builtin_amdgcn_wmma_f32_16x16x32_bf16(false, aF, false, bf1,
                                                       (short)0, acc1, false, false);
        acc2 = __builtin_amdgcn_wmma_f32_16x16x32_bf16(false, aF, false, bf2,
                                                       (short)0, acc2, false, false);
        acc3 = __builtin_amdgcn_wmma_f32_16x16x32_bf16(false, aF, false, bf3,
                                                       (short)0, acc3, false, false);
    }

    // C/D layout: VGPR i -> row (tile_m + lhalf*8 + i), col (n0 + t*16 + lmod)
    // out is streamed once -> non-temporal stores (protect A's L2 residency).
    const int row0 = tile_m + lhalf * 8;
    v8f accs[4] = {acc0, acc1, acc2, acc3};
#pragma unroll
    for (int t = 0; t < 4; ++t) {
        const int n  = n0 + t * 16 + lmod;
        const float bv = bias[n];
#pragma unroll
        for (int i = 0; i < 8; ++i) {
            __builtin_nontemporal_store(accs[t][i] + bv,
                                        &out[(size_t)(row0 + i) * C_ + n]);
        }
    }
}

// ---------------------------------------------------------------------------
// Launch: inputs are grid (fp32), W (fp32 [512,256]), b (fp32 [256]).
// Workspace layout: [0, 256KB) Wt bf16; [256KB, 256KB + 268MB) A bf16.
// ---------------------------------------------------------------------------
extern "C" void kernel_launch(void* const* d_in, const int* in_sizes, int n_in,
                              void* d_out, int out_size, void* d_ws, size_t ws_size,
                              hipStream_t stream) {
    const float* grid = (const float*)d_in[0];
    const float* Wsrc = (const float*)d_in[1];
    const float* bias = (const float*)d_in[2];
    float* out = (float*)d_out;

    uint16_t* Wt = (uint16_t*)d_ws;
    uint16_t* A  = (uint16_t*)((char*)d_ws + (size_t)C_ * K_ * sizeof(uint16_t));

    // 1) repack weights to transposed bf16
    repack_w_kernel<<<K_, C_, 0, stream>>>(Wsrc, Wt);

    // 2) both cummax scans -> bf16 activation matrix A [M, 512]
    dim3 gcm(B_ * H_, 2);   // H_ == W_ so both scan modes use B*128 blocks
    cummax_kernel<<<gcm, C_, 0, stream>>>(grid, A);

    // 3) WMMA GEMM + bias (B staged in LDS, shared by all 8 waves)
    dim3 ggemm(M_ / 128, C_ / NTILE_);
    wmma_gemm_kernel<<<ggemm, 256, 0, stream>>>(A, Wt, bias, out);
}
